// VectorQuantizerLayer_27204322852880
// MI455X (gfx1250) — matrix operand order repaired and checked by reading, not compiled, not run on hardware
//
#include <hip/hip_runtime.h>

typedef __attribute__((ext_vector_type(16))) _Float16 v16h;
typedef __attribute__((ext_vector_type(8)))  float    v8f;

#define VQ_D 128
#define VQ_K 1024
#define VQ_N (32 * 4096)

// ---------------------------------------------------------------------------
// Workspace layout (bytes):
//   0        CTf32  [K][D] f32   (524288)  codebook transposed, f32 (for gather)
//   524288   CT16   [K][D] f16   (262144)  codebook transposed, f16 (for WMMA B)
//   786432   norms  [K]    f32   (4096)    ||c_k||^2
//   790528   idx    [N]    i32   (524288)  argmin indices
//   1314816  lossAcc       f32   (4)
// ---------------------------------------------------------------------------

// Kernel 1: transpose codebook, build f16 copy, norms, zero loss accumulator.
__global__ void vq_prep(const float* __restrict__ qv,      // [D][K] row-major
                        float* __restrict__ ctf,           // [K][D]
                        _Float16* __restrict__ ct16,       // [K][D]
                        float* __restrict__ norms,         // [K]
                        float* __restrict__ lossAcc) {
    const int k = blockIdx.x;    // code index 0..1023
    const int d = threadIdx.x;   // dim index  0..127
    const float v = qv[(size_t)d * VQ_K + k];
    ctf[k * VQ_D + d]  = v;
    ct16[k * VQ_D + d] = (_Float16)v;

    float s = v * v;
    #pragma unroll
    for (int off = 16; off; off >>= 1) s += __shfl_down(s, off, 32);
    __shared__ float ls[4];
    if ((threadIdx.x & 31) == 0) ls[threadIdx.x >> 5] = s;
    __syncthreads();
    if (threadIdx.x == 0) {
        norms[k] = ls[0] + ls[1] + ls[2] + ls[3];
        if (k == 0) *lossAcc = 0.0f;
    }
}

// Kernel 2: WMMA distance GEMM + argmin.
// Each wave owns 32 rows of X (two 16-row A tiles resident in VGPRs) and
// sweeps all 1024 codes in 64 chunks of 16 columns. B fragments are shared by
// both A tiles (8 wmma per 8 b128 loads) and double-buffered so chunk c+1's
// loads overlap chunk c's WMMAs.  dist = ||c||^2 - 2*(x.c)  (||x||^2 dropped).
__global__ __launch_bounds__(256)
void vq_argmin(const float* __restrict__ X,        // [N][D] f32
               const _Float16* __restrict__ ct16,  // [K][D] f16
               const float* __restrict__ norms,    // [K]
               int* __restrict__ idxOut) {         // [N]
    const int lane = threadIdx.x & 31;
    const int wave = threadIdx.x >> 5;
    const int nlo  = lane & 15;
    const int hi   = lane >> 4;
    const int rowBase = (blockIdx.x * 8 + wave) * 32;

    // A fragment gather in contiguous k order (same as B, not AMD's table order):
    // lane m holds row m; elements 0..15 -> K = k0..k0+15,
    // k0 = 32*frag + 16*(lane>=16).
    v16h a[2][4];
    #pragma unroll
    for (int t = 0; t < 2; ++t) {
        const float* xr = X + (size_t)(rowBase + 16 * t + nlo) * VQ_D;
        #pragma unroll
        for (int f = 0; f < 4; ++f) {
            const int k0 = 32 * f + 16 * hi;
            #pragma unroll
            for (int e = 0; e < 8; ++e) a[t][f][e]     = (_Float16)xr[k0 + e];
            #pragma unroll
            for (int e = 0; e < 8; ++e) a[t][f][8 + e] = (_Float16)xr[k0 + 8 + e];
        }
    }

    float best[2][8];
    int   bidx[2][8];
    #pragma unroll
    for (int t = 0; t < 2; ++t)
        #pragma unroll
        for (int v = 0; v < 8; ++v) { best[t][v] = 3.0e38f; bidx[t][v] = 0; }

    // Preload chunk 0 B fragments (lane holds column lane&15; 16 consecutive
    // K values at offset 16*(lane>=16) inside each 32-deep fragment -> 32B).
    v16h b0[4], b1[4];
    float n0, n1;
    {
        const _Float16* br = ct16 + (size_t)nlo * VQ_D + 16 * hi;
        #pragma unroll
        for (int f = 0; f < 4; ++f) b0[f] = *(const v16h*)(br + 32 * f);
        n0 = norms[nlo];
    }

    #pragma unroll 2
    for (int c = 0; c < 64; ++c) {
        // Prefetch next chunk (wrapped index -> branchless; extra re-read of
        // chunk 0 on the last iteration is harmless and cache-hot).
        const int cn = (c + 1) & 63;
        const _Float16* brn = ct16 + (size_t)(cn * 16 + nlo) * VQ_D + 16 * hi;
        #pragma unroll
        for (int f = 0; f < 4; ++f) b1[f] = *(const v16h*)(brn + 32 * f);
        n1 = norms[cn * 16 + nlo];

        v8f acc0 = {}, acc1 = {};
        #pragma unroll
        for (int f = 0; f < 4; ++f) {
            acc0 = __builtin_amdgcn_wmma_f32_16x16x32_f16(
                false, a[0][f], false, b0[f], (short)0, acc0, false, false);
            acc1 = __builtin_amdgcn_wmma_f32_16x16x32_f16(
                false, a[1][f], false, b0[f], (short)0, acc1, false, false);
        }

        const int col = c * 16 + nlo;
        #pragma unroll
        for (int v = 0; v < 8; ++v) {
            const float d0 = n0 - 2.0f * acc0[v];
            if (d0 < best[0][v]) { best[0][v] = d0; bidx[0][v] = col; }
            const float d1 = n0 - 2.0f * acc1[v];
            if (d1 < best[1][v]) { best[1][v] = d1; bidx[1][v] = col; }
        }

        // Rotate double buffers (eliminated by the unroll-by-2).
        #pragma unroll
        for (int f = 0; f < 4; ++f) b0[f] = b1[f];
        n0 = n1;
    }

    // Min-reduce across the 16 lanes of each half-wave (columns of this row),
    // tie-break to lowest index to match jnp.argmin.
    #pragma unroll
    for (int t = 0; t < 2; ++t)
        #pragma unroll
        for (int v = 0; v < 8; ++v) {
            #pragma unroll
            for (int off = 1; off < 16; off <<= 1) {
                const float ob = __shfl_xor(best[t][v], off, 32);
                const int   oi = __shfl_xor(bidx[t][v], off, 32);
                if (ob < best[t][v] || (ob == best[t][v] && oi < bidx[t][v])) {
                    best[t][v] = ob; bidx[t][v] = oi;
                }
            }
        }
    if (nlo == 0) {
        #pragma unroll
        for (int t = 0; t < 2; ++t)
            #pragma unroll
            for (int v = 0; v < 8; ++v)
                idxOut[rowBase + 16 * t + 8 * hi + v] = bidx[t][v];
    }
}

// Kernel 3: gather winning f32 codebook rows to output + accumulate
// sum((q - x)^2). 8 rows per block, one wave per row, float4 per lane.
__global__ __launch_bounds__(256)
void vq_gather_loss(const float* __restrict__ X,
                    const float* __restrict__ ctf,
                    const int* __restrict__ idx,
                    float* __restrict__ out,
                    float* __restrict__ lossAcc) {
    const int lane = threadIdx.x & 31;
    const int rw   = threadIdx.x >> 5;        // 0..7
    const int r    = blockIdx.x * 8 + rw;
    const int code = idx[r];
    const float4 q = *(const float4*)(ctf + (size_t)code * VQ_D + lane * 4);
    const float4 x = *(const float4*)(X   + (size_t)r    * VQ_D + lane * 4);
    *(float4*)(out + (size_t)r * VQ_D + lane * 4) = q;

    const float dx = q.x - x.x, dy = q.y - x.y, dz = q.z - x.z, dw = q.w - x.w;
    float s = dx * dx + dy * dy + dz * dz + dw * dw;
    #pragma unroll
    for (int off = 16; off; off >>= 1) s += __shfl_down(s, off, 32);
    __shared__ float ls[8];
    if (lane == 0) ls[rw] = s;
    __syncthreads();
    if (threadIdx.x == 0) {
        float t = 0.0f;
        #pragma unroll
        for (int i = 0; i < 8; ++i) t += ls[i];
        atomicAdd(lossAcc, t);
    }
}

// Kernel 4: loss = (BETA + 1) * mean = 1.25 * sum / (N*D)
__global__ void vq_finalize(const float* __restrict__ lossAcc,
                            float* __restrict__ outLoss) {
    outLoss[0] = 1.25f * lossAcc[0] / (float)(VQ_N * VQ_D);
}

extern "C" void kernel_launch(void* const* d_in, const int* in_sizes, int n_in,
                              void* d_out, int out_size, void* d_ws, size_t ws_size,
                              hipStream_t stream) {
    (void)in_sizes; (void)n_in; (void)out_size; (void)ws_size;
    const float* X  = (const float*)d_in[0];   // [32,4096,128] f32
    const float* QV = (const float*)d_in[1];   // [128,1024]    f32
    float* out = (float*)d_out;                // [N*D] quantized_st + [1] loss

    char* ws = (char*)d_ws;
    float*    ctf   = (float*)(ws);
    _Float16* ct16  = (_Float16*)(ws + 524288);
    float*    norms = (float*)(ws + 786432);
    int*      idx   = (int*)(ws + 790528);
    float*    lossA = (float*)(ws + 1314816);

    vq_prep<<<VQ_K, VQ_D, 0, stream>>>(QV, ctf, ct16, norms, lossA);
    vq_argmin<<<VQ_N / 256, 256, 0, stream>>>(X, ct16, norms, idx);
    vq_gather_loss<<<VQ_N / 8, 256, 0, stream>>>(X, ctf, idx, out, lossA);
    vq_finalize<<<1, 1, 0, stream>>>(lossA, out + (size_t)VQ_N * VQ_D);
}
